// NGCFDL_60473139527901
// MI455X (gfx1250) — matrix operand order repaired and checked by reading, not compiled, not run on hardware
//
#include <hip/hip_runtime.h>

#define NUSERS 20000
#define NITEMS 30000
#define NN     50000          // total rows
#define DD     768
#define NE     400000
#define MTILES 3125           // NN/16
#define KTILES 24             // DD/32
#define NTILES 48             // DD/16
#define NTPW   6              // n-tiles per wave (48 / 8 waves)

typedef __attribute__((ext_vector_type(16))) __bf16 v16bf;
typedef __attribute__((ext_vector_type(8)))  float  v8f;

__device__ __forceinline__ float lrelu(float x) { return x > 0.0f ? x : 0.01f * x; }

// ---------------------------------------------------------------------------
// Init: ego = concat(user_emb, item_emb); also write out[:, 0:768] (layer-0 slice)
// ---------------------------------------------------------------------------
__global__ void k_init(const float* __restrict__ ue, const float* __restrict__ ie,
                       float* __restrict__ ego, float* __restrict__ out) {
  int gid = blockIdx.x * blockDim.x + threadIdx.x;   // over NN*192 float4s
  if (gid >= NN * 192) return;
  int row = gid / 192, c4 = gid % 192;
  float4 v = (row < NUSERS)
      ? ((const float4*)(ue + (size_t)row * DD))[c4]
      : ((const float4*)(ie + (size_t)(row - NUSERS) * DD))[c4];
  ((float4*)(ego + (size_t)row * DD))[c4] = v;
  ((float4*)(out + (size_t)row * (3 * DD)))[c4] = v;
}

__global__ void k_zero(float* __restrict__ p, int n4) {
  int gid = blockIdx.x * blockDim.x + threadIdx.x;
  if (gid < n4) ((float4*)p)[gid] = make_float4(0.f, 0.f, 0.f, 0.f);
}

// ---------------------------------------------------------------------------
// COO SpMM scatter: side[rows[e]] += vals[e] * ego[cols[e]]  (float atomics)
// ---------------------------------------------------------------------------
__global__ void k_spmm(const int* __restrict__ rows, const int* __restrict__ cols,
                       const float* __restrict__ vals, const float* __restrict__ ego,
                       float* __restrict__ side) {
  long gid = (long)blockIdx.x * blockDim.x + threadIdx.x;   // NE*192 float4s
  if (gid >= (long)NE * 192) return;
  int e = (int)(gid / 192), c = (int)(gid % 192);
  int r = rows[e], cl = cols[e];
  float v = vals[e];
  float4 x = ((const float4*)(ego + (size_t)cl * DD))[c];
  float* dst = side + (size_t)r * DD + c * 4;
  atomicAdd(dst + 0, v * x.x);
  atomicAdd(dst + 1, v * x.y);
  atomicAdd(dst + 2, v * x.z);
  atomicAdd(dst + 3, v * x.w);
}

// ---------------------------------------------------------------------------
// Repack side / ego*side (fp32 row-major) into WMMA 16x32 bf16 A-fragments.
// A layout (16-bit, wave32): M = lane%16; lanes>=16 add K-offset 8;
//   pair p<4 -> K = koff+2p(+lo); p>=4 -> K = 16+koff+2(p-4)(+lo).
// Block b = mt*KTILES+kt; element = lane*16 + j (contiguous 32B per lane).
// ---------------------------------------------------------------------------
__global__ void k_frag_a(const float* __restrict__ side, const float* __restrict__ ego,
                         __bf16* __restrict__ sideA, __bf16* __restrict__ egosideA) {
  int gid = blockIdx.x * blockDim.x + threadIdx.x;   // MTILES*KTILES*32
  if (gid >= MTILES * KTILES * 32) return;
  int lane = gid & 31, tile = gid >> 5;
  int mt = tile / KTILES, kt = tile % KTILES;
  int row  = mt * 16 + (lane & 15);
  int k0   = kt * 32;
  int koff = (lane >> 4) * 8;
  size_t base = (size_t)tile * 512 + lane * 16;
  const float* srow = side + (size_t)row * DD;
  const float* erow = ego  + (size_t)row * DD;
#pragma unroll
  for (int p = 0; p < 8; ++p) {
    int K = k0 + ((p < 4) ? (koff + 2 * p) : (16 + koff + 2 * (p - 4)));
    float2 s = *(const float2*)(srow + K);
    float2 e = *(const float2*)(erow + K);
    sideA[base + 2 * p]     = (__bf16)s.x;
    sideA[base + 2 * p + 1] = (__bf16)s.y;
    egosideA[base + 2 * p]     = (__bf16)(e.x * s.x);
    egosideA[base + 2 * p + 1] = (__bf16)(e.y * s.y);
  }
}

// ---------------------------------------------------------------------------
// Repack a [768,768] fp32 weight into WMMA 32x16 bf16 B-fragments.
// B layout (16-bit, wave32): N = lane%16; lanes>=16 add K-offset 8;
// j<8 -> K = koff+j, j>=8 -> K = 16+koff+(j-8); per-lane 16 contiguous bf16.
// Block b = kt*NTILES+nt.
// ---------------------------------------------------------------------------
__global__ void k_frag_b(const float* __restrict__ W, __bf16* __restrict__ WB) {
  int gid = blockIdx.x * blockDim.x + threadIdx.x;   // KTILES*NTILES*32
  if (gid >= KTILES * NTILES * 32) return;
  int lane = gid & 31, tile = gid >> 5;
  int kt = tile / NTILES, nt = tile % NTILES;
  int kbase = kt * 32 + (lane >> 4) * 8;
  int n = nt * 16 + (lane & 15);
  size_t base = (size_t)tile * 512 + lane * 16;
#pragma unroll
  for (int j = 0; j < 16; ++j)
    WB[base + j] = (__bf16)W[(size_t)(kbase + (j < 8 ? j : j + 8)) * DD + n];
}

// ---------------------------------------------------------------------------
// Fused dual GEMM + bias + leaky-relu:
//   ego = lrelu(side@gc_w + gc_b) + lrelu((ego*side)@bi_w + bi_b)
// One workgroup (8 waves) per 16-row m-tile; wave w covers n-tiles
// {w, w+8, ..., w+40} so A fragments are loaded once per wave and reused
// across 6 n-tiles (x2 GEMMs) in registers; all 48 n-tiles of an m-tile
// are temporally co-resident for L2/WGP$ locality.
// ---------------------------------------------------------------------------
__global__ void __launch_bounds__(256)
k_gemm(const __bf16* __restrict__ sideA, const __bf16* __restrict__ egosideA,
       const __bf16* __restrict__ wgc, const __bf16* __restrict__ wbi,
       const float* __restrict__ gcb, const float* __restrict__ bib,
       float* __restrict__ ego) {
  int mt   = blockIdx.x;
  int wave = threadIdx.x >> 5;
  int lane = threadIdx.x & 31;

  v8f acc1[NTPW];
  v8f acc2[NTPW];
#pragma unroll
  for (int j = 0; j < NTPW; ++j) { acc1[j] = {}; acc2[j] = {}; }

  for (int kt = 0; kt < KTILES; ++kt) {
    size_t aOff = ((size_t)(mt * KTILES + kt)) * 512 + lane * 16;
    v16bf a1 = *(const v16bf*)(sideA + aOff);
    v16bf a2 = *(const v16bf*)(egosideA + aOff);
#pragma unroll
    for (int j = 0; j < NTPW; ++j) {
      int nt = wave + 8 * j;
      size_t bOff = ((size_t)(kt * NTILES + nt)) * 512 + lane * 16;
      v16bf b1 = *(const v16bf*)(wgc + bOff);
      v16bf b2 = *(const v16bf*)(wbi + bOff);
      acc1[j] = __builtin_amdgcn_wmma_f32_16x16x32_bf16(false, a1, false, b1,
                                                        (short)0, acc1[j], false, false);
      acc2[j] = __builtin_amdgcn_wmma_f32_16x16x32_bf16(false, a2, false, b2,
                                                        (short)0, acc2[j], false, false);
    }
  }

  // C/D layout: VGPR r -> M = r + 8*(lane/16), N = lane%16
  int mbase = mt * 16 + (lane >> 4) * 8;
#pragma unroll
  for (int j = 0; j < NTPW; ++j) {
    int n = (wave + 8 * j) * 16 + (lane & 15);
    float c1 = gcb[n], c2 = bib[n];
#pragma unroll
    for (int r = 0; r < 8; ++r) {
      float v = lrelu(acc1[j][r] + c1) + lrelu(acc2[j][r] + c2);
      ego[(size_t)(mbase + r) * DD + n] = v;
    }
  }
}

// ---------------------------------------------------------------------------
// Row-wise L2 normalize into output slice; one 256-thread block per row.
// ---------------------------------------------------------------------------
__global__ void k_norm(const float* __restrict__ ego, float* __restrict__ out, int colOff) {
  int row = blockIdx.x, tid = threadIdx.x;
  const float* e = ego + (size_t)row * DD;
  float v0 = e[tid], v1 = e[tid + 256], v2 = e[tid + 512];
  __shared__ float red[256];
  red[tid] = v0 * v0 + v1 * v1 + v2 * v2;
  __syncthreads();
  for (int off = 128; off > 0; off >>= 1) {
    if (tid < off) red[tid] += red[tid + off];
    __syncthreads();
  }
  float scale = 1.0f / fmaxf(sqrtf(red[0]), 1e-12f);
  float* o = out + (size_t)row * (3 * DD) + colOff;
  o[tid]       = v0 * scale;
  o[tid + 256] = v1 * scale;
  o[tid + 512] = v2 * scale;
}

// ---------------------------------------------------------------------------
extern "C" void kernel_launch(void* const* d_in, const int* in_sizes, int n_in,
                              void* d_out, int out_size, void* d_ws, size_t ws_size,
                              hipStream_t stream) {
  const float* user_emb = (const float*)d_in[0];
  const float* item_emb = (const float*)d_in[1];
  const float* gc_w     = (const float*)d_in[2];   // [2,768,768]
  const float* gc_b     = (const float*)d_in[3];   // [2,768]
  const float* bi_w     = (const float*)d_in[4];
  const float* bi_b     = (const float*)d_in[5];
  const int*   rows     = (const int*)d_in[6];
  const int*   cols     = (const int*)d_in[7];
  const float* vals     = (const float*)d_in[8];
  float* out = (float*)d_out;

  // workspace carve-up (~463 MB total)
  size_t nd = (size_t)NN * DD;
  float*  ego      = (float*)d_ws;
  float*  side     = ego + nd;
  __bf16* sideA    = (__bf16*)(side + nd);
  __bf16* egosideA = sideA + nd;
  __bf16* wgc      = egosideA + nd;
  __bf16* wbi      = wgc + (size_t)DD * DD;

  dim3 blk(256);
  int g_init = (NN * 192 + 255) / 256;                     // 37500
  int g_spmm = (int)(((long)NE * 192 + 255) / 256);        // 300000
  int g_fra  = (MTILES * KTILES * 32 + 255) / 256;         // 9375
  int g_frb  = (KTILES * NTILES * 32 + 255) / 256;         // 144

  k_init<<<g_init, blk, 0, stream>>>(user_emb, item_emb, ego, out);

  for (int i = 0; i < 2; ++i) {
    k_zero<<<g_init, blk, 0, stream>>>(side, NN * 192);
    k_spmm<<<g_spmm, blk, 0, stream>>>(rows, cols, vals, ego, side);
    k_frag_a<<<g_fra, blk, 0, stream>>>(side, ego, sideA, egosideA);
    k_frag_b<<<g_frb, blk, 0, stream>>>(gc_w + (size_t)i * DD * DD, wgc);
    k_frag_b<<<g_frb, blk, 0, stream>>>(bi_w + (size_t)i * DD * DD, wbi);
    k_gemm<<<MTILES, blk, 0, stream>>>(sideA, egosideA, wgc, wbi,
                                       gc_b + (size_t)i * DD,
                                       bi_b + (size_t)i * DD, ego);
    k_norm<<<NN, blk, 0, stream>>>(ego, out, (i + 1) * DD);
  }
}
